// PretrainedMambaVisionClassifier_87677462381071
// MI455X (gfx1250) — matrix operand order repaired and checked by reading, not compiled, not run on hardware
//
#include <hip/hip_runtime.h>

// ---------------- model constants ----------------
static constexpr int kNL   = 4;
static constexpr int kDM   = 1024;
static constexpr int kDI   = 2048;
static constexpr int kDS   = 16;
static constexpr int kDR   = 64;
static constexpr int kDC   = 4;
static constexpr int kB    = 8;
static constexpr int kL    = 224;
static constexpr int kDLOC = 672;
static constexpr int kNC   = 10;
static constexpr int kXDB  = 128;      // padded x_dbl row stride (96 -> 128)
static constexpr float kEPS = 1e-5f;

typedef __attribute__((ext_vector_type(16))) __bf16        v16bf;
typedef __attribute__((ext_vector_type(8)))  float         v8f;
typedef __attribute__((ext_vector_type(4)))  unsigned int  v4u;

union FragCast { v4u u[2]; v16bf v; };

__device__ __forceinline__ unsigned pack_bf16(float lo, float hi) {
    union { __bf16 h[2]; unsigned u; } p;
    p.h[0] = (__bf16)lo;            // low 16 bits = even K (ISA pair order)
    p.h[1] = (__bf16)hi;
    return p.u;
}

__device__ __forceinline__ float softplus_f(float r) {
    return (r > 20.f) ? r : log1pf(__expf(r));
}

// =====================================================================
// FAST WMMA GEMM:  C[M,N] = act( A*B [+ bias] [+ Cin] )
// Requires M%64==0, N%128==0, K%32==0 (lda/ldb/ldc multiples of 4).
// Block: 256 threads (8 waves), tile 64x128, K-step 32.
// Double-buffered LDS of packed-bf16 K-pairs + register prefetch of the
// next tile: global b128 loads for step k+32 are issued before the
// ds_load/WMMA burst of step k, hiding VMEM latency under matrix math.
// One s_barrier per K-step.
// Layout: Ap[row][kp], BpT[col][kp]; stride 20 dwords -> 20*r mod 64 is
// distinct over 16 consecutive rows => conflict-free b128 fragment reads.
// Wave w: rows (w>>1)*16, cols (w&1)*64 (+j*16); 4 accums reuse 1 A frag.
// =====================================================================
template<bool BIAS, bool CIN, bool SP>
__global__ __launch_bounds__(256)
void wmma_gemm_fast(const float* __restrict__ A, const float* __restrict__ Bw,
                    const float* __restrict__ bias, const float* __restrict__ Cin,
                    float* __restrict__ C, int K, int lda, int ldb, int ldc)
{
    __shared__ unsigned Ap[2][64][20];     // 2 x (64 rows x 16 K-pairs +4 pad)
    __shared__ unsigned BpT[2][128][20];   // 2 x (128 cols x 16 K-pairs +4 pad)

    const int tid  = threadIdx.x;
    const int lane = tid & 31;
    const int wave = tid >> 5;
    const int waveRow = wave >> 1;         // 0..3
    const int waveCol = wave & 1;          // 0..1
    const int half = lane >> 4;            // K-half selector
    const int lr   = lane & 15;

    const int m0 = blockIdx.y * 64;
    const int n0 = blockIdx.x * 128;

    // per-thread staging coordinates (A: row/f4, B: kp/col-group)
    const int aRow0 = tid >> 3,  aF4 = tid & 7;          // + 32 rows on iter 1
    const int bKp   = tid & 15,  bCg = tid >> 4;         // + 16 col-groups on iter 1

    v8f acc[4] = {};

    float4 aReg[2], bReg0[2], bReg1[2];
    auto load_tile = [&](int k0) {
        #pragma unroll
        for (int it = 0; it < 2; ++it) {
            aReg[it] = *(const float4*)&A[(size_t)(m0 + aRow0 + it * 32) * lda + k0 + aF4 * 4];
            const int col = (bCg + it * 16) * 4;
            bReg0[it] = *(const float4*)&Bw[(size_t)(k0 + 2 * bKp)     * ldb + n0 + col];
            bReg1[it] = *(const float4*)&Bw[(size_t)(k0 + 2 * bKp + 1) * ldb + n0 + col];
        }
    };

    load_tile(0);                          // prologue prefetch

    int buf = 0;
    for (int k0 = 0; k0 < K; k0 += 32, buf ^= 1) {
        // ---- pack prefetched regs into LDS buffer `buf` ----
        #pragma unroll
        for (int it = 0; it < 2; ++it) {
            unsigned* dst = &Ap[buf][aRow0 + it * 32][aF4 * 2];
            dst[0] = pack_bf16(aReg[it].x, aReg[it].y);
            dst[1] = pack_bf16(aReg[it].z, aReg[it].w);
            const int col = (bCg + it * 16) * 4;
            BpT[buf][col + 0][bKp] = pack_bf16(bReg0[it].x, bReg1[it].x);
            BpT[buf][col + 1][bKp] = pack_bf16(bReg0[it].y, bReg1[it].y);
            BpT[buf][col + 2][bKp] = pack_bf16(bReg0[it].z, bReg1[it].z);
            BpT[buf][col + 3][bKp] = pack_bf16(bReg0[it].w, bReg1[it].w);
        }
        __syncthreads();

        // ---- prefetch next tile into registers (flies under the WMMAs) ----
        if (k0 + 32 < K) load_tile(k0 + 32);

        // ---- fragments: pure b128 LDS loads, already bf16-packed ----
        FragCast af;
        {
            const int ar = waveRow * 16 + lr;
            const int h4 = half * 4;
            af.u[0] = *(const v4u*)&Ap[buf][ar][h4];       // K = kb..kb+7
            af.u[1] = *(const v4u*)&Ap[buf][ar][h4 + 8];   // K = kb+16..kb+23
        }
    const int hb = half * 4;
        #pragma unroll
        for (int j = 0; j < 4; ++j) {
            FragCast bfr;
            const int c = waveCol * 64 + j * 16 + lr;
            bfr.u[0] = *(const v4u*)&BpT[buf][c][hb];
            bfr.u[1] = *(const v4u*)&BpT[buf][c][hb + 8];
            acc[j] = __builtin_amdgcn_wmma_f32_16x16x32_bf16(false, af.v, false, bfr.v,
                                                             (short)0, acc[j], false, false);
        }
    }

    // ---- branch-free epilogue: lane half h, reg v -> M = h*8+v, N = lr ----
    #pragma unroll
    for (int j = 0; j < 4; ++j) {
        const int cb = n0 + waveCol * 64 + j * 16 + lr;
        float bv = 0.f;
        if constexpr (BIAS) bv = bias[cb];
        #pragma unroll
        for (int v = 0; v < 8; ++v) {
            const int row = m0 + waveRow * 16 + half * 8 + v;
            float r = acc[j][v] + bv;
            if constexpr (CIN) r += Cin[(size_t)row * ldc + cb];
            if constexpr (SP)  r = softplus_f(r);
            C[(size_t)row * ldc + cb] = r;
        }
    }
}

// =====================================================================
// Zero-pad w_x[l] (DI x 96) into (DI x 128) so the x_dbl GEMM can use the
// fast unguarded WMMA path (padding cols of B are zero -> outputs zero).
// =====================================================================
__global__ __launch_bounds__(256)
void pad_wx_kernel(const float* __restrict__ src, float* __restrict__ dst)
{
    const int i = blockIdx.x * 256 + threadIdx.x;        // over DI*128
    if (i >= kDI * kXDB) return;
    const int r = i >> 7, c = i & 127;
    dst[i] = (c < 96) ? src[r * 96 + c] : 0.f;
}

// =====================================================================
// RMSNorm over kDM=1024 features; one block per row (strided rows).
// =====================================================================
__global__ __launch_bounds__(256)
void rmsnorm_kernel(const float* __restrict__ in, const float* __restrict__ w,
                    float* __restrict__ out, long strideIn, long strideOut)
{
    __shared__ float red[256];
    const int tid = threadIdx.x;
    const float* r = in + (long)blockIdx.x * strideIn;
    float s = 0.f;
    for (int i = tid; i < kDM; i += 256) { float v = r[i]; s += v * v; }
    red[tid] = s;
    __syncthreads();
    for (int off = 128; off > 0; off >>= 1) {
        if (tid < off) red[tid] += red[tid + off];
        __syncthreads();
    }
    const float scale = rsqrtf(red[0] / (float)kDM + kEPS);
    float* o = out + (long)blockIdx.x * strideOut;
    for (int i = tid; i < kDM; i += 256) o[i] = r[i] * scale * w[i];
}

// =====================================================================
// Depthwise causal conv (DC=4) + SiLU.
// =====================================================================
__global__ __launch_bounds__(256)
void conv_silu_kernel(const float* __restrict__ xz, const float* __restrict__ wconv,
                      const float* __restrict__ bconv, float* __restrict__ xc)
{
    const long idx = (long)blockIdx.x * 256 + threadIdx.x;
    if (idx >= (long)kB * kL * kDI) return;
    const int d  = (int)(idx % kDI);
    const long bt = idx / kDI;
    const int t  = (int)(bt % kL);
    const int b  = (int)(bt / kL);
    float acc = bconv[d];
    #pragma unroll
    for (int k = 0; k < kDC; ++k) {
        const int ts = t - (kDC - 1) + k;
        if (ts >= 0)
            acc += xz[((long)(b * kL + ts)) * (2 * kDI) + d] * wconv[d * kDC + k];
    }
    xc[idx] = acc / (1.f + __expf(-acc));   // silu
}

// =====================================================================
// Selective scan: thread per (b,d), 16-wide state in regs, B_t/C_t via LDS.
// Fuses  y = (scan + Dp*u) * silu(z).   grid = (B, DI/256)
// xdb has padded row stride kXDB=128: dtr cols 0..63, Bm 64..79, Cm 80..95.
// =====================================================================
__global__ __launch_bounds__(256)
void scan_kernel(const float* __restrict__ xdb, const float* __restrict__ dt,
                 const float* __restrict__ xc, const float* __restrict__ xz,
                 const float* __restrict__ A_log, const float* __restrict__ Dp,
                 float* __restrict__ y)
{
    const int b = blockIdx.x;
    const int d = blockIdx.y * 256 + threadIdx.x;

    float Arow[kDS];
    #pragma unroll
    for (int s = 0; s < kDS; ++s) Arow[s] = -__expf(A_log[d * kDS + s]);
    const float Dv = Dp[d];

    float hs[kDS];
    #pragma unroll
    for (int s = 0; s < kDS; ++s) hs[s] = 0.f;

    __shared__ float Bsh[kDS];
    __shared__ float Csh[kDS];

    for (int t = 0; t < kL; ++t) {
        const long base = (long)(b * kL + t);
        if (threadIdx.x < 32) {
            const int s = threadIdx.x & 15;
            const float v = xdb[base * kXDB + kDR + (threadIdx.x >> 4) * kDS + s];
            if (threadIdx.x < 16) Bsh[s] = v; else Csh[s] = v;
        }
        __syncthreads();

        const float dtv = dt[base * kDI + d];
        const float u   = xc[base * kDI + d];
        const float du  = dtv * u;
        float acc = 0.f;
        #pragma unroll
        for (int s = 0; s < kDS; ++s) {
            hs[s] = __expf(dtv * Arow[s]) * hs[s] + du * Bsh[s];
            acc += hs[s] * Csh[s];
        }
        const float z  = xz[base * (2 * kDI) + kDI + d];
        const float sz = z / (1.f + __expf(-z));
        y[base * kDI + d] = (acc + Dv * u) * sz;
        __syncthreads();
    }
}

// =====================================================================
// Final FC: out[8,10] = pooled_norm @ w_fc + b_fc
// =====================================================================
__global__ __launch_bounds__(128)
void fc_kernel(const float* __restrict__ pn, const float* __restrict__ w_fc,
               const float* __restrict__ b_fc, float* __restrict__ out)
{
    const int i = threadIdx.x;
    if (i >= kB * kNC) return;
    const int b = i / kNC, c = i % kNC;
    float acc = b_fc[c];
    for (int k = 0; k < kDM; ++k) acc += pn[b * kDM + k] * w_fc[k * kNC + c];
    out[i] = acc;
}

// =====================================================================
extern "C" void kernel_launch(void* const* d_in, const int* in_sizes, int n_in,
                              void* d_out, int out_size, void* d_ws, size_t ws_size,
                              hipStream_t stream)
{
    (void)in_sizes; (void)n_in; (void)out_size; (void)ws_size;

    const float* x      = (const float*)d_in[0];
    const float* w_proj = (const float*)d_in[1];
    const float* b_proj = (const float*)d_in[2];
    const float* norm_w = (const float*)d_in[3];
    const float* w_in   = (const float*)d_in[4];
    const float* w_conv = (const float*)d_in[5];
    const float* b_conv = (const float*)d_in[6];
    const float* w_x    = (const float*)d_in[7];
    const float* w_dt   = (const float*)d_in[8];
    const float* b_dt   = (const float*)d_in[9];
    const float* A_log  = (const float*)d_in[10];
    const float* Dp     = (const float*)d_in[11];
    const float* w_out  = (const float*)d_in[12];
    const float* fnw    = (const float*)d_in[13];
    const float* w_fc   = (const float*)d_in[14];
    const float* b_fc   = (const float*)d_in[15];

    const int M = kB * kL;   // 1792

    float* ws  = (float*)d_ws;
    float* h   = ws;                            // M*DM
    float* xn  = h   + (size_t)M * kDM;         // M*DM
    float* xz  = xn  + (size_t)M * kDM;         // M*2DI
    float* xc  = xz  + (size_t)M * 2 * kDI;     // M*DI
    float* xdb = xc  + (size_t)M * kDI;         // M*kXDB (padded 128)
    float* dt  = xdb + (size_t)M * kXDB;        // M*DI
    float* yb  = dt  + (size_t)M * kDI;         // M*DI
    float* pn  = yb  + (size_t)M * kDI;         // 8*DM
    float* wxp = pn  + (size_t)kB * kDM;        // DI*kXDB (padded w_x[l])

    const dim3 blk(256);

    // h = x @ w_proj + b_proj
    wmma_gemm_fast<true, false, false><<<dim3(kDM / 128, M / 64), blk, 0, stream>>>(
        x, w_proj, b_proj, nullptr, h, kDLOC, kDLOC, kDM, kDM);

    for (int l = 0; l < kNL; ++l) {
        rmsnorm_kernel<<<M, blk, 0, stream>>>(h, norm_w + (size_t)l * kDM, xn,
                                              kDM, kDM);
        // xz = xn @ w_in[l]
        wmma_gemm_fast<false, false, false><<<dim3(2 * kDI / 128, M / 64), blk, 0, stream>>>(
            xn, w_in + (size_t)l * kDM * 2 * kDI, nullptr, nullptr, xz,
            kDM, kDM, 2 * kDI, 2 * kDI);
        // xc = silu(conv(xp) + b_conv)
        {
            const long total = (long)M * kDI;
            conv_silu_kernel<<<(unsigned)((total + 255) / 256), blk, 0, stream>>>(
                xz, w_conv + (size_t)l * kDI * kDC, b_conv + (size_t)l * kDI, xc);
        }
        // pad w_x[l] (DI x 96) -> wxp (DI x 128) so x_dbl uses the fast path
        pad_wx_kernel<<<(kDI * kXDB + 255) / 256, blk, 0, stream>>>(
            w_x + (size_t)l * kDI * 96, wxp);
        // x_dbl(padded) = xc @ wxp     (N=128, cols 96..127 are zeros)
        wmma_gemm_fast<false, false, false><<<dim3(kXDB / 128, M / 64), blk, 0, stream>>>(
            xc, wxp, nullptr, nullptr, xdb, kDI, kDI, kXDB, kXDB);
        // dt = softplus(x_dbl[:, :64] @ w_dt[l] + b_dt[l])   (lda = padded 128)
        wmma_gemm_fast<true, false, true><<<dim3(kDI / 128, M / 64), blk, 0, stream>>>(
            xdb, w_dt + (size_t)l * kDR * kDI, b_dt + (size_t)l * kDI, nullptr, dt,
            kDR, kXDB, kDI, kDI);
        // selective scan (+Dp*u, *silu(z))
        scan_kernel<<<dim3(kB, kDI / 256), blk, 0, stream>>>(
            xdb, dt, xc, xz, A_log + (size_t)l * kDI * kDS, Dp + (size_t)l * kDI, yb);
        // h += y @ w_out[l]  (residual via Cin, in place)
        wmma_gemm_fast<false, true, false><<<dim3(kDM / 128, M / 64), blk, 0, stream>>>(
            yb, w_out + (size_t)l * kDI * kDM, nullptr, h, h,
            kDI, kDI, kDM, kDM);
    }

    rmsnorm_kernel<<<kB, blk, 0, stream>>>(h + (size_t)(kL - 1) * kDM, fnw, pn,
                                           (long)kL * kDM, kDM);
    fc_kernel<<<1, 128, 0, stream>>>(pn, w_fc, b_fc, (float*)d_out);
}
